// TimeAwareNanoGPTTrajectoryForecaster_61357902790676
// MI455X (gfx1250) — matrix operand order repaired and checked by reading, not compiled, not run on hardware
//
#include <hip/hip_runtime.h>
#include <stdint.h>
#include <stddef.h>

typedef _Float16 f16;
typedef __attribute__((ext_vector_type(16))) _Float16 v16h;
typedef __attribute__((ext_vector_type(8)))  float    v8f;
typedef unsigned int u32;
typedef unsigned long long u64;
typedef __attribute__((ext_vector_type(4))) unsigned int u32x4;
typedef __attribute__((ext_vector_type(8))) int          i32x8;
typedef __attribute__((ext_vector_type(4))) int          i32x4;

#define B_   64
#define S_   512
#define H_   32
#define D_   512
#define NH_  8
#define HD_  64
#define NL_  6
#define FF_  2048
#define M_   256
#define R_   4
#define BS_  (B_ * S_)
#define TDLEN_ (S_ + H_ - 1)   // 543

#define WMMA_F32_F16(a, b, c) \
  __builtin_amdgcn_wmma_f32_16x16x32_f16(false, (a), false, (b), (short)0, (c), false, false)

#if defined(__has_builtin)
#if __has_builtin(__builtin_amdgcn_tensor_load_to_lds) && __has_builtin(__builtin_amdgcn_s_wait_tensorcnt)
#define HAVE_TDM 1
#endif
#endif
#ifndef HAVE_TDM
#define HAVE_TDM 0
#endif

// ---------------------------------------------------------------------------
// Fragment loaders (wave32; A and B use contiguous k = 16*(lane>>4) + s)
// ---------------------------------------------------------------------------
static __device__ __forceinline__ v16h load_a_frag(const f16* sm, int rowBase, int ldk, int kBase) {
  const int lane = threadIdx.x & 31;
  const f16* rowp = sm + (size_t)(rowBase + (lane & 15)) * ldk + kBase;
  const int kg = (lane >> 4) * 8;
  union { v16h v; unsigned u[8]; } r;
#pragma unroll
  for (int i = 0; i < 8; ++i) {
    int k0 = 2 * kg + i * 2;
    r.u[i] = *(const unsigned*)(rowp + k0);
  }
  return r.v;
}

static __device__ __forceinline__ v16h load_bT_frag(const f16* sm, int nBase, int ldk, int kBase) {
  const int lane = threadIdx.x & 31;
  return *(const v16h*)(sm + (size_t)(nBase + (lane & 15)) * ldk + kBase + (lane >> 4) * 16);
}

static __device__ __forceinline__ float redmax16(float v) {
#pragma unroll
  for (int m = 1; m <= 8; m <<= 1) v = fmaxf(v, __shfl_xor(v, m, 32));
  return v;
}
static __device__ __forceinline__ float redadd16(float v) {
#pragma unroll
  for (int m = 1; m <= 8; m <<= 1) v += __shfl_xor(v, m, 32);
  return v;
}

// ---------------------------------------------------------------------------
// Tensor Data Mover: 2D f16 tile -> LDS (ISA 8.3/8.4 descriptor layout)
// 6-arg builtin variant: (u32x4 g0, i32x8 g1, i32x4 g2, i32x4 g3, i32x8 g4, i32 cpol)
// ---------------------------------------------------------------------------
#if HAVE_TDM
static __device__ __forceinline__ void tdm_load_tile_f16(
    const f16* gsrc, unsigned ldsByteAddr,
    int rowHalfs, int rows, int remRows, int remHalfs, int strideHalfs)
{
  u64 ga   = (u64)(uintptr_t)gsrc;
  u32 td0  = (u32)(remHalfs >> 2);     // tensor_dim0 in 8B units
  u32 td1  = (u32)remRows;             // tensor_dim1 (rows; OOB rows read as zero)
  u32 tl0  = (u32)(rowHalfs >> 2);     // tile_dim0 in 8B units
  u32 tl1  = (u32)rows;                // tile_dim1
  u64 st0  = (u64)(strideHalfs >> 2);  // tensor_dim0_stride in 8B units

  u32x4 g0;
  g0[0] = 1u;                                               // count=1, user mode
  g0[1] = ldsByteAddr;                                      // lds_addr
  g0[2] = (u32)ga;                                          // global_addr[31:0]
  g0[3] = ((u32)(ga >> 32) & 0x01FFFFFFu) | (2u << 30);     // addr[56:32] | type=2

  i32x8 g1;
  g1[0] = (int)(3u << 16);                                  // data_size = 8B
  g1[1] = (int)((td0 & 0xFFFFu) << 16);                     // tensor_dim0[15:0]
  g1[2] = (int)((td0 >> 16) | ((td1 & 0xFFFFu) << 16));     // td0[31:16] | td1[15:0]
  g1[3] = (int)((td1 >> 16) | (tl0 << 16));                 // td1[31:16] | tile_dim0
  g1[4] = (int)(tl1 & 0xFFFFu);                             // tile_dim1 (tile_dim2=0)
  g1[5] = (int)(u32)st0;                                    // stride0[31:0]
  g1[6] = (int)((u32)(st0 >> 32) & 0xFFFFu);                // stride0[47:32]
  g1[7] = 0;
  i32x4 gz4 = {0, 0, 0, 0};
  i32x8 gz8 = {0, 0, 0, 0, 0, 0, 0, 0};
  __builtin_amdgcn_tensor_load_to_lds(g0, g1, gz4, gz4, gz8, 0);
}
#endif

// ---------------------------------------------------------------------------
// WMMA GEMM:  C[M,N] = epi( A16[M,K] @ W16[N,K]^T + bias (+res) )
// N % 64 == 0, K % 64 == 0; M bounds handled (TDM OOB / predication).
// ---------------------------------------------------------------------------
#define GBM 128
#define GBN 64
#define GBK 64

static __device__ __forceinline__ void gemm_step(const f16* tA, const f16* tB,
                                                 int wm, int wn, v8f acc[2][2]) {
  v16h a00 = load_a_frag(tA, wm,      GBK, 0);
  v16h a01 = load_a_frag(tA, wm,      GBK, 32);
  v16h a10 = load_a_frag(tA, wm + 16, GBK, 0);
  v16h a11 = load_a_frag(tA, wm + 16, GBK, 32);
  v16h b00 = load_bT_frag(tB, wn,      GBK, 0);
  v16h b01 = load_bT_frag(tB, wn,      GBK, 32);
  v16h b10 = load_bT_frag(tB, wn + 16, GBK, 0);
  v16h b11 = load_bT_frag(tB, wn + 16, GBK, 32);
  acc[0][0] = WMMA_F32_F16(a00, b00, acc[0][0]);
  acc[0][0] = WMMA_F32_F16(a01, b01, acc[0][0]);
  acc[0][1] = WMMA_F32_F16(a00, b10, acc[0][1]);
  acc[0][1] = WMMA_F32_F16(a01, b11, acc[0][1]);
  acc[1][0] = WMMA_F32_F16(a10, b00, acc[1][0]);
  acc[1][0] = WMMA_F32_F16(a11, b01, acc[1][0]);
  acc[1][1] = WMMA_F32_F16(a10, b10, acc[1][1]);
  acc[1][1] = WMMA_F32_F16(a11, b11, acc[1][1]);
}

__global__ __launch_bounds__(256)
void gemm_wmma_kernel(const f16* __restrict__ A,
                      const f16* __restrict__ W,
                      const float* __restrict__ bias,
                      const float* __restrict__ res,
                      void* __restrict__ Cv,
                      int M, int N, int K, int actGelu, int outF16)
{
  __shared__ __align__(32) f16 smA[2][GBM * GBK];   // 2 x 16KB
  __shared__ __align__(32) f16 smB[2][GBN * GBK];   // 2 x  8KB

  const int tid  = threadIdx.x;
  const int lane = tid & 31;
  const int wave = tid >> 5;
  const int wm   = (wave >> 1) * 32;
  const int wn   = (wave & 1) * 32;
  const int blockM = blockIdx.x * GBM;
  const int blockN = blockIdx.y * GBN;

  const v8f vzero = {0.f, 0.f, 0.f, 0.f, 0.f, 0.f, 0.f, 0.f};
  v8f acc[2][2];
  acc[0][0] = vzero; acc[0][1] = vzero; acc[1][0] = vzero; acc[1][1] = vzero;

  const int nk = K / GBK;

#if HAVE_TDM
  // ---- Tensor-Data-Mover staged, double-buffered pipeline ----
  const f16* Ab = A + (size_t)blockM * K;
  const f16* Wb = W + (size_t)blockN * K;
  const unsigned ldsA[2] = { (unsigned)(uintptr_t)&smA[0][0], (unsigned)(uintptr_t)&smA[1][0] };
  const unsigned ldsB[2] = { (unsigned)(uintptr_t)&smB[0][0], (unsigned)(uintptr_t)&smB[1][0] };
  const int remM = M - blockM;
  const int remN = N - blockN;

  if (wave == 0) {
    tdm_load_tile_f16(Ab, ldsA[0], GBK, GBM, remM, K, K);
    tdm_load_tile_f16(Wb, ldsB[0], GBK, GBN, remN, K, K);
    __builtin_amdgcn_s_wait_tensorcnt((short)0);
  }
  __syncthreads();
  for (int kt = 0; kt < nk; ++kt) {
    const int cur = kt & 1;
    if (wave == 0 && (kt + 1) < nk) {
      const int k0n = (kt + 1) * GBK;
      tdm_load_tile_f16(Ab + k0n, ldsA[cur ^ 1], GBK, GBM, remM, K - k0n, K);
      tdm_load_tile_f16(Wb + k0n, ldsB[cur ^ 1], GBK, GBN, remN, K - k0n, K);
    }
    gemm_step(&smA[cur][0], &smB[cur][0], wm, wn, acc);
    if (wave == 0 && (kt + 1) < nk) __builtin_amdgcn_s_wait_tensorcnt((short)0);
    __syncthreads();
  }
#else
  // ---- Fallback: manual f16 staging, single-buffered ----
  for (int kt = 0; kt < nk; ++kt) {
    const int k0 = kt * GBK;
    __syncthreads();
#pragma unroll
    for (int i = 0; i < (GBM * GBK) / (256 * 8); ++i) {    // 4 iters of uint4
      int e8 = i * 256 + tid;
      int row = e8 >> 3;
      int c8 = (e8 & 7) * 8;
      int gr = blockM + row;
      uint4 v = {0u, 0u, 0u, 0u};
      if (gr < M) v = *(const uint4*)(A + (size_t)gr * K + k0 + c8);
      *(uint4*)(&smA[0][row * GBK + c8]) = v;
    }
#pragma unroll
    for (int i = 0; i < (GBN * GBK) / (256 * 8); ++i) {    // 2 iters
      int e8 = i * 256 + tid;
      int row = e8 >> 3;
      int c8 = (e8 & 7) * 8;
      *(uint4*)(&smB[0][row * GBK + c8]) =
          *(const uint4*)(W + (size_t)(blockN + row) * K + k0 + c8);
    }
    __syncthreads();
    gemm_step(&smA[0][0], &smB[0][0], wm, wn, acc);
  }
#endif

  // Epilogue: C/D layout -> row = (lane>>4)*8 + v, col = lane&15.
  float* Cf = (float*)Cv;
  f16*   Ch = (f16*)Cv;
  const int colL = lane & 15;
  const int mr0  = (lane >> 4) * 8;
#pragma unroll
  for (int i = 0; i < 2; ++i) {
#pragma unroll
    for (int j = 0; j < 2; ++j) {
      int n = blockN + wn + j * 16 + colL;
      int mBase = blockM + wm + i * 16;
      float bv = bias ? bias[n] : 0.f;
#pragma unroll
      for (int v = 0; v < 8; ++v) {
        int m = mBase + mr0 + v;
        if (m < M) {
          float val = acc[i][j][v] + bv;
          if (actGelu) val = 0.5f * val * (1.f + erff(val * 0.70710678118f));
          if (res) val += res[(size_t)m * N + n];
          if (outF16) Ch[(size_t)m * N + n] = (f16)val;
          else        Cf[(size_t)m * N + n] = val;
        }
      }
    }
  }
}

// ---------------------------------------------------------------------------
// Flash attention, f16 qkv in / f16 out. One wave per (b, head, 16-query tile).
// ---------------------------------------------------------------------------
__global__ __launch_bounds__(32)
void attn_kernel(const f16* __restrict__ qkv, f16* __restrict__ o)
{
  __shared__ __align__(32) f16 sq[16 * 64];
  __shared__ __align__(32) f16 sk[32 * 64];
  __shared__ __align__(32) f16 sv[64 * 32];   // transposed [d][key]
  __shared__ __align__(32) f16 sp[16 * 32];

  const int lane = threadIdx.x;
  const int qt = blockIdx.x;
  const int hd = blockIdx.y;
  const int b  = blockIdx.z;

  const f16* qbase = qkv + ((size_t)(b * S_ + qt * 16)) * (3 * D_) + hd * HD_;
#pragma unroll
  for (int i = 0; i < 4; ++i) {
    int e8 = lane + i * 32;
    int row = e8 >> 3, c8 = (e8 & 7) * 8;
    *(uint4*)(sq + row * 64 + c8) = *(const uint4*)(qbase + (size_t)row * (3 * D_) + c8);
  }
  __syncthreads();
  asm volatile("s_wait_dscnt 0" ::: "memory");
  v16h aq0 = load_a_frag(sq, 0, 64, 0);
  v16h aq1 = load_a_frag(sq, 0, 64, 32);

  float mmax[8], lsum[8];
  const v8f vzero = {0.f, 0.f, 0.f, 0.f, 0.f, 0.f, 0.f, 0.f};
  v8f oacc[4];
#pragma unroll
  for (int v = 0; v < 8; ++v) { mmax[v] = -3.0e38f; lsum[v] = 0.f; }
#pragma unroll
  for (int t = 0; t < 4; ++t) oacc[t] = vzero;

  const int colL = lane & 15;
  const int mr0  = (lane >> 4) * 8;
  const int nchunk = ((qt * 16 + 15) >> 5) + 1;

  for (int c = 0; c < nchunk; ++c) {
    __syncthreads();
    const f16* kbase = qkv + ((size_t)(b * S_ + c * 32)) * (3 * D_) + D_ + hd * HD_;
    const f16* vbase = kbase + D_;
#pragma unroll
    for (int i = 0; i < 8; ++i) {
      int e8 = lane + i * 32;
      int row = e8 >> 3, c8 = (e8 & 7) * 8;
      *(uint4*)(sk + row * 64 + c8) = *(const uint4*)(kbase + (size_t)row * (3 * D_) + c8);
      union { uint4 q; f16 h[8]; } uv;
      uv.q = *(const uint4*)(vbase + (size_t)row * (3 * D_) + c8);
#pragma unroll
      for (int t = 0; t < 8; ++t) sv[(c8 + t) * 32 + row] = uv.h[t];
    }
    __syncthreads();
    asm volatile("s_wait_dscnt 0" ::: "memory");

#pragma unroll
    for (int j = 0; j < 2; ++j) {
      v8f s = vzero;
      v16h bk0 = load_bT_frag(sk, j * 16, 64, 0);
      v16h bk1 = load_bT_frag(sk, j * 16, 64, 32);
      s = WMMA_F32_F16(aq0, bk0, s);
      s = WMMA_F32_F16(aq1, bk1, s);
      int key = c * 32 + j * 16 + colL;
#pragma unroll
      for (int v = 0; v < 8; ++v) {
        int qrow = qt * 16 + mr0 + v;
        float sc = (key <= qrow) ? (s[v] * 0.125f) : -1.0e9f;   // 1/sqrt(HD) on scores
        float mnew = fmaxf(mmax[v], redmax16(sc));
        float rescale = __expf(mmax[v] - mnew);
        float p = __expf(sc - mnew);
        lsum[v] = lsum[v] * rescale + redadd16(p);
        mmax[v] = mnew;
#pragma unroll
        for (int t = 0; t < 4; ++t) oacc[t][v] *= rescale;
        sp[(mr0 + v) * 32 + j * 16 + colL] = (f16)p;
      }
    }
    __syncthreads();
    asm volatile("s_wait_dscnt 0" ::: "memory");
    v16h ap = load_a_frag(sp, 0, 32, 0);
#pragma unroll
    for (int t = 0; t < 4; ++t) {
      v16h bv = load_bT_frag(sv, t * 16, 32, 0);
      oacc[t] = WMMA_F32_F16(ap, bv, oacc[t]);
    }
  }

  f16* obase = o + ((size_t)(b * S_ + qt * 16)) * D_ + hd * HD_;
#pragma unroll
  for (int t = 0; t < 4; ++t)
#pragma unroll
    for (int v = 0; v < 8; ++v)
      obase[(size_t)(mr0 + v) * D_ + t * 16 + colL] = (f16)(oacc[t][v] / lsum[v]);
}

// ---------------------------------------------------------------------------
// Small kernels
// ---------------------------------------------------------------------------
__global__ __launch_bounds__(256)
void cvt_f16_kernel(const float* __restrict__ src, f16* __restrict__ dst, int n)
{
  int i = blockIdx.x * 256 + threadIdx.x;
  if (i < n) dst[i] = (f16)src[i];
}

__global__ __launch_bounds__(256)
void embed_kernel(const float* __restrict__ wte, const float* __restrict__ wpe,
                  const float* __restrict__ iuv, const float* __restrict__ ctr,
                  const float* __restrict__ td, float* __restrict__ x)
{
  int idx = blockIdx.x * 256 + threadIdx.x;
  if (idx >= BS_ * D_) return;
  int d = idx & (D_ - 1);
  int s = (idx >> 9) & (S_ - 1);
  int b = idx >> 18;
  float f0 = iuv[(size_t)(b * S_ + s) * 2 + 0];
  float f1 = iuv[(size_t)(b * S_ + s) * 2 + 1];
  float f2 = ctr[b * 2 + 0];
  float f3 = ctr[b * 2 + 1];
  float f4 = td[(size_t)b * TDLEN_ + s];
  const float* w = wte + (size_t)d * 5;
  x[idx] = w[0] * f0 + w[1] * f1 + w[2] * f2 + w[3] * f3 + w[4] * f4 + wpe[(size_t)s * D_ + d];
}

__global__ __launch_bounds__(256)
void ln_kernel(const float* __restrict__ x, const float* __restrict__ w, f16* __restrict__ y)
{
  __shared__ float rs[8], rs2[8];
  size_t row = blockIdx.x;
  const float* xr = x + row * D_;
  float s = 0.f, s2 = 0.f;
  for (int i = threadIdx.x; i < D_; i += 256) { float v = xr[i]; s += v; s2 += v * v; }
#pragma unroll
  for (int m = 1; m <= 16; m <<= 1) { s += __shfl_xor(s, m, 32); s2 += __shfl_xor(s2, m, 32); }
  if ((threadIdx.x & 31) == 0) { rs[threadIdx.x >> 5] = s; rs2[threadIdx.x >> 5] = s2; }
  __syncthreads();
  float ts = 0.f, ts2 = 0.f;
#pragma unroll
  for (int i = 0; i < 8; ++i) { ts += rs[i]; ts2 += rs2[i]; }
  float mu = ts * (1.f / D_);
  float var = ts2 * (1.f / D_) - mu * mu;
  float rstd = rsqrtf(var + 1e-5f);
  f16* yr = y + row * D_;
  for (int i = threadIdx.x; i < D_; i += 256) yr[i] = (f16)((xr[i] - mu) * rstd * w[i]);
}

__global__ __launch_bounds__(256)
void gather_kernel(const f16* __restrict__ x, const int* __restrict__ pos, f16* __restrict__ hid)
{
  int idx = blockIdx.x * 256 + threadIdx.x;
  if (idx >= B_ * D_) return;
  int b = idx >> 9, d = idx & (D_ - 1);
  hid[idx] = x[((size_t)b * S_ + pos[b]) * D_ + d];
}

__global__ __launch_bounds__(256)
void timecum_kernel(const int* __restrict__ pos, const float* __restrict__ td, float* __restrict__ tb)
{
  int idx = blockIdx.x * 256 + threadIdx.x;
  if (idx >= B_ * H_) return;
  int b = idx >> 5, h = idx & (H_ - 1);
  int start = pos[b] + 1;
  float cum = 0.f;
  for (int j = 0; j < h; ++j) cum += td[(size_t)b * TDLEN_ + start + j];
  tb[idx] = fminf(fmaxf(cum * (1.f / 3600.f), 0.f), 2.f);
}

__global__ __launch_bounds__(256)
void t1_kernel(const float* __restrict__ tb, const float* __restrict__ t1w,
               const float* __restrict__ t1b, f16* __restrict__ u1)
{
  int idx = blockIdx.x * 256 + threadIdx.x;
  if (idx >= B_ * H_ * 256) return;
  int j = idx & 255, bh = idx >> 8;
  float v = tb[bh] * t1w[j] + t1b[j];
  u1[idx] = (f16)fmaxf(v, 0.f);
}

__global__ __launch_bounds__(256)
void concat_kernel(const float* __restrict__ base, const float* __restrict__ tr, f16* __restrict__ cc)
{
  int idx = blockIdx.x * 256 + threadIdx.x;
  if (idx >= B_ * H_ * 768) return;
  int c = idx % 768, bh = idx / 768;
  int b = bh >> 5;
  float v = (c < D_) ? base[(size_t)b * D_ + c] : tr[(size_t)bh * 256 + (c - D_)];
  cc[idx] = (f16)v;
}

__global__ __launch_bounds__(256)
void fourier_kernel(const float* __restrict__ cu, const float* __restrict__ cv,
                    const float* __restrict__ tuv, float* __restrict__ out)
{
  int bh = blockIdx.x * 256 + threadIdx.x;
  if (bh >= B_ * H_) return;
  const float PI = 3.14159265358979f;
  float tu = tuv[(size_t)bh * 2 + 0];
  float tv = tuv[(size_t)bh * 2 + 1];
  const float* up = cu + (size_t)bh * (M_ * R_ * 2);
  const float* vp = cv + (size_t)bh * (M_ * R_ * 2);

  float aR[4] = {0}, aI[4] = {0}, bR[4] = {0}, bI[4] = {0};
  float guR[16] = {0}, guI[16] = {0}, gvR[16] = {0}, gvI[16] = {0};

  for (int m = 0; m < M_; ++m) {
    float su, cuu, sv_, cvv;
    __sincosf(PI * (float)m * tu, &su, &cuu);
    __sincosf(PI * (float)m * tv, &sv_, &cvv);
    float UR[4], UI[4], VR[4], VI[4];
#pragma unroll
    for (int r = 0; r < 4; ++r) {
      UR[r] = up[(size_t)(m * R_ + r) * 2 + 0];
      UI[r] = up[(size_t)(m * R_ + r) * 2 + 1];
      VR[r] = vp[(size_t)(m * R_ + r) * 2 + 0];
      VI[r] = vp[(size_t)(m * R_ + r) * 2 + 1];
    }
#pragma unroll
    for (int r = 0; r < 4; ++r) {
      aR[r] += cuu * UR[r] - su * UI[r];
      aI[r] += cuu * UI[r] + su * UR[r];
      bR[r] += cvv * VR[r] + sv_ * VI[r];
      bI[r] += sv_ * VR[r] - cvv * VI[r];
    }
#pragma unroll
    for (int r = 0; r < 4; ++r)
#pragma unroll
      for (int s = 0; s < 4; ++s) {
        guR[r * 4 + s] += UR[r] * UR[s] + UI[r] * UI[s];
        guI[r * 4 + s] += UR[r] * UI[s] - UI[r] * UR[s];
        gvR[r * 4 + s] += VR[r] * VR[s] + VI[r] * VI[s];
        gvI[r * 4 + s] += VR[r] * VI[s] - VI[r] * VR[s];
      }
  }
  float psiR = 0.f, psiI = 0.f;
#pragma unroll
  for (int r = 0; r < 4; ++r) {
    psiR += aR[r] * bR[r] - aI[r] * bI[r];
    psiI += aR[r] * bI[r] + aI[r] * bR[r];
  }
  float Z = 0.f;
#pragma unroll
  for (int r = 0; r < 4; ++r)
#pragma unroll
    for (int s = 0; s < 4; ++s)
      Z += guR[r * 4 + s] * gvR[s * 4 + r] - guI[r * 4 + s] * gvI[s * 4 + r];
  Z *= 4.0f;
  float pdf = (psiR * psiR + psiI * psiI) / Z;
  out[bh] = logf(pdf);
}

// ---------------------------------------------------------------------------
// Host orchestration
// ---------------------------------------------------------------------------
static void gemm(hipStream_t st, const f16* A, const f16* W, const float* bias,
                 const float* res, void* C, int M, int N, int K, int gelu, int outF16)
{
  dim3 grid((M + GBM - 1) / GBM, N / GBN);
  gemm_wmma_kernel<<<grid, 256, 0, st>>>(A, W, bias, res, C, M, N, K, gelu, outF16);
}
static void cvt(hipStream_t st, const float* src, f16* dst, int n)
{
  cvt_f16_kernel<<<(n + 255) / 256, 256, 0, st>>>(src, dst, n);
}

extern "C" void kernel_launch(void* const* d_in, const int* in_sizes, int n_in,
                              void* d_out, int out_size, void* d_ws, size_t ws_size,
                              hipStream_t stream)
{
  (void)in_sizes; (void)n_in; (void)out_size; (void)ws_size;

  const float* wte = (const float*)d_in[0];
  const float* wpe = (const float*)d_in[1];
  const float *ln1[NL_], *qkvw[NL_], *projw[NL_], *ln2[NL_], *fcw[NL_], *fcpw[NL_];
  for (int l = 0; l < NL_; ++l) {
    int base = 2 + l * 6;
    ln1[l]  = (const float*)d_in[base + 0];
    qkvw[l] = (const float*)d_in[base + 1];
    projw[l]= (const float*)d_in[base + 2];
    ln2[l]  = (const float*)d_in[base + 3];
    fcw[l]  = (const float*)d_in[base + 4];
    fcpw[l] = (const float*)d_in[base + 5];
  }
  const float* lnf   = (const float*)d_in[38];
  const float* out_w = (const float*)d_in[39];
  const float* out_b = (const float*)d_in[40];
  const float* t1_w  = (const float*)d_in[41];
  const float* t1_b  = (const float*)d_in[42];
  const float* t2_w  = (const float*)d_in[43];
  const float* t2_b  = (const float*)d_in[44];
  const float* cmb_w = (const float*)d_in[45];
  const float* cmb_b = (const float*)d_in[46];
  const float* fu_w  = (const float*)d_in[47];
  const float* fu_b  = (const float*)d_in[48];
  const float* fv_w  = (const float*)d_in[49];
  const float* fv_b  = (const float*)d_in[50];
  const float* input_uv  = (const float*)d_in[51];
  const float* centre_ll = (const float*)d_in[52];
  const float* target_uv = (const float*)d_in[53];
  const int*   cpos      = (const int*)d_in[54];
  const float* tdel      = (const float*)d_in[55];
  float* outp = (float*)d_out;

  // ---- workspace carve (256B aligned bumps) ----
  char* wsb = (char*)d_ws;
  auto alloc = [&](size_t bytes) -> char* {
    char* r = wsb;
    wsb += (bytes + 255) & ~(size_t)255;
    return r;
  };
  const int QKV_SZ = 3 * D_ * D_, PROJ_SZ = D_ * D_, FC_SZ = FF_ * D_, FCP_SZ = FF_ * D_;
  const int OUTW_SZ = D_ * D_, T2W_SZ = 256 * 256, CMBW_SZ = D_ * 768;
  const int FUW_SZ = 2 * M_ * R_ * D_;

  float* x     = (float*)alloc((size_t)BS_ * D_ * 4);
  f16*   h16   = (f16*)  alloc((size_t)BS_ * D_ * 2);
  f16*   o16   = (f16*)  alloc((size_t)BS_ * D_ * 2);
  f16*   big16 = (f16*)  alloc((size_t)BS_ * FF_ * 2);   // qkv16 / ffh16, later head scratch
  f16*   qkv16[NL_], *proj16[NL_], *fc16[NL_], *fcp16[NL_];
  for (int l = 0; l < NL_; ++l) {
    qkv16[l]  = (f16*)alloc((size_t)QKV_SZ * 2);
    proj16[l] = (f16*)alloc((size_t)PROJ_SZ * 2);
    fc16[l]   = (f16*)alloc((size_t)FC_SZ * 2);
    fcp16[l]  = (f16*)alloc((size_t)FCP_SZ * 2);
  }
  f16* outw16 = (f16*)alloc((size_t)OUTW_SZ * 2);
  f16* t2w16  = (f16*)alloc((size_t)T2W_SZ * 2);
  f16* cmbw16 = (f16*)alloc((size_t)CMBW_SZ * 2);
  f16* fuw16  = (f16*)alloc((size_t)FUW_SZ * 2);
  f16* fvw16  = (f16*)alloc((size_t)FUW_SZ * 2);

  // head-stage buffers alias into big16 (free after the layer loop)
  char* hbb = (char*)big16;
  auto halloc = [&](size_t bytes) -> char* {
    char* r = hbb;
    hbb += (bytes + 255) & ~(size_t)255;
    return r;
  };
  f16*   hidden16 = (f16*)  halloc((size_t)B_ * D_ * 2);
  float* baseo    = (float*)halloc((size_t)B_ * D_ * 4);
  float* tb       = (float*)halloc((size_t)B_ * H_ * 4);
  f16*   u116     = (f16*)  halloc((size_t)B_ * H_ * 256 * 2);
  float* tr       = (float*)halloc((size_t)B_ * H_ * 256 * 4);
  f16*   cc16     = (f16*)  halloc((size_t)B_ * H_ * 768 * 2);
  f16*   rep16    = (f16*)  halloc((size_t)B_ * H_ * D_ * 2);
  float* cu       = (float*)halloc((size_t)B_ * H_ * 2 * M_ * R_ * 4);
  float* cv       = (float*)halloc((size_t)B_ * H_ * 2 * M_ * R_ * 4);

  // ---- one-time (per launch) weight conversion to f16 ----
  for (int l = 0; l < NL_; ++l) {
    cvt(stream, qkvw[l], qkv16[l], QKV_SZ);
    cvt(stream, projw[l], proj16[l], PROJ_SZ);
    cvt(stream, fcw[l], fc16[l], FC_SZ);
    cvt(stream, fcpw[l], fcp16[l], FCP_SZ);
  }
  cvt(stream, out_w, outw16, OUTW_SZ);
  cvt(stream, t2_w, t2w16, T2W_SZ);
  cvt(stream, cmb_w, cmbw16, CMBW_SZ);
  cvt(stream, fu_w, fuw16, FUW_SZ);
  cvt(stream, fv_w, fvw16, FUW_SZ);

  // ---- embedding ----
  embed_kernel<<<(BS_ * D_) / 256, 256, 0, stream>>>(wte, wpe, input_uv, centre_ll, tdel, x);

  // ---- transformer layers ----
  for (int l = 0; l < NL_; ++l) {
    ln_kernel<<<BS_, 256, 0, stream>>>(x, ln1[l], h16);
    gemm(stream, h16, qkv16[l], nullptr, nullptr, big16, BS_, 3 * D_, D_, 0, 1);
    attn_kernel<<<dim3(S_ / 16, NH_, B_), 32, 0, stream>>>(big16, o16);
    gemm(stream, o16, proj16[l], nullptr, x, x, BS_, D_, D_, 0, 0);
    ln_kernel<<<BS_, 256, 0, stream>>>(x, ln2[l], h16);
    gemm(stream, h16, fc16[l], nullptr, nullptr, big16, BS_, FF_, D_, 1, 1);   // GELU, f16 out
    gemm(stream, big16, fcp16[l], nullptr, x, x, BS_, D_, FF_, 0, 0);
  }

  // ---- head ----
  ln_kernel<<<BS_, 256, 0, stream>>>(x, lnf, h16);
  gather_kernel<<<(B_ * D_) / 256, 256, 0, stream>>>(h16, cpos, hidden16);
  gemm(stream, hidden16, outw16, out_b, nullptr, baseo, B_, D_, D_, 0, 0);

  timecum_kernel<<<(B_ * H_) / 256, 256, 0, stream>>>(cpos, tdel, tb);
  t1_kernel<<<(B_ * H_ * 256) / 256, 256, 0, stream>>>(tb, t1_w, t1_b, u116);
  gemm(stream, u116, t2w16, t2_b, nullptr, tr, B_ * H_, 256, 256, 0, 0);

  concat_kernel<<<(B_ * H_ * 768) / 256, 256, 0, stream>>>(baseo, tr, cc16);
  gemm(stream, cc16, cmbw16, cmb_b, nullptr, rep16, B_ * H_, D_, 768, 0, 1);
  gemm(stream, rep16, fuw16, fu_b, nullptr, cu, B_ * H_, 2 * M_ * R_, D_, 0, 0);
  gemm(stream, rep16, fvw16, fv_b, nullptr, cv, B_ * H_, 2 * M_ * R_, D_, 0, 0);

  fourier_kernel<<<(B_ * H_) / 256, 256, 0, stream>>>(cu, cv, target_uv, outp);
}
